// RQL_63170378990082
// MI455X (gfx1250) — compile-verified, matrix operand order checked by reading, not run on hardware
//
#include <hip/hip_runtime.h>
#include <hip/hip_bf16.h>
#include <stdint.h>

// ---------------- constants ----------------
#define SRC_LEN 50
#define TRG_LEN 50
#define BATCH   64
#define EN_V    16000
#define SPA_V   24000
#define EMB     256
#define HID     256
#define TSTEPS  100
#define NOUT    24003          // SPA_V + 3
#define NPAD    24016          // padded to 16
#define EN_EOS  2
#define EN_NULL 3
#define EN_PAD  1
#define SPA_EOS 2
#define SPA_NULL 3
#define SPA_PAD 1

// LDS staging geometry for k_out (row stride padded to 528B = 132 dwords:
// consecutive rows shift 4 banks -> conflict-free 16-lane b128 reads)
#define LDS_ROW_B   528
#define LDS_ROW_E   264            // elements (bf16)
#define LDS_B_BASE  (64 * LDS_ROW_B)          // 33792
#define LDS_TOTAL   (LDS_B_BASE + 32 * LDS_ROW_B)  // 50688 bytes

typedef __attribute__((ext_vector_type(16))) __bf16    v16bf;
typedef __attribute__((ext_vector_type(8)))  float     v8f;
typedef __attribute__((ext_vector_type(8)))  unsigned  v8u;

__device__ __forceinline__ unsigned short f2bf_rne(float f) {
  unsigned u = __builtin_bit_cast(unsigned, f);
  return (unsigned short)((u + 0x7FFFu + ((u >> 16) & 1u)) >> 16);
}

// Load a 16-bf16 WMMA fragment for one lane.
// 16-bit A/B layout: lane half=0 -> K = {0..7, 16..23}; half=1 -> K = {8..15, 24..31},
// packed as dword pairs (K even, K odd).
__device__ __forceinline__ v16bf load_frag(const unsigned short* p, int half) {
  v8u u;
#pragma unroll
  for (int pr = 0; pr < 8; ++pr) {
    int kb = (pr < 4) ? (half * 8 + 2 * pr) : (8 + half * 8 + 2 * pr);
    u[pr] = *(const unsigned*)(p + kb);
  }
  return __builtin_bit_cast(v16bf, u);
}

__device__ __forceinline__ v8f wmma_bf16(v16bf a, v16bf b, v8f c) {
  return __builtin_amdgcn_wmma_f32_16x16x32_bf16(false, a, false, b, (short)0, c,
                                                 false, false);
}

// Async global->LDS 16B copy (ASYNCcnt-tracked; CDNA5 GLOBAL_LOAD_ASYNC_TO_LDS_B128)
__device__ __forceinline__ void async_b128(unsigned lds_off, unsigned long long gaddr) {
  asm volatile("global_load_async_to_lds_b128 %0, %1, off"
               :: "v"(lds_off), "v"(gaddr) : "memory");
}

// ---------------- one-time kernels ----------------
__global__ void k_zero(float* p, int n) {
  int i = blockIdx.x * 256 + threadIdx.x;
  if (i < n) p[i] = 0.0f;
}

__global__ void k_init(float* h, int* wo, int* wr, int* na, int* te, int* ia,
                       int* ja, int* counts) {
  int tid = blockIdx.x * 256 + threadIdx.x;
  if (tid < BATCH * HID) h[tid] = 0.0f;
  if (tid < BATCH) {
    wo[tid] = SPA_NULL; wr[tid] = 0; na[tid] = 0; te[tid] = 0;
    ia[tid] = 0; ja[tid] = 0;
  }
  if (tid < 3) counts[tid] = 0;
}

// WcatT[n][kk], n in [0,768), kk in [0,768): kk<512 -> W_x[kk][n], else W_h[kk-512][n]
__global__ void k_convW(const float* Wx, const float* Wh, unsigned short* WcatT) {
  int idx = blockIdx.x * 256 + threadIdx.x;
  if (idx >= 768 * 768) return;
  int n = idx / 768, kk = idx % 768;
  float v = (kk < 512) ? Wx[kk * 768 + n] : Wh[(kk - 512) * 768 + n];
  WcatT[n * 768 + kk] = f2bf_rne(v);
}

// WoutT[n][k], n in [0,NPAD), k in [0,256): n<NOUT -> W_out[k][n], else 0
__global__ void k_convWout(const float* Wout, unsigned short* WoutT) {
  int idx = blockIdx.x * 256 + threadIdx.x;
  if (idx >= NPAD * 256) return;
  int n = idx / 256, k = idx % 256;
  float v = (n < NOUT) ? Wout[k * NOUT + n] : 0.0f;
  WoutT[n * 256 + k] = f2bf_rne(v);
}

// ---------------- per-step kernels ----------------
// One workgroup, 8 waves: build x=[E_en|E_spa|h] in bf16, two WMMA GEMMs
// (gx: K=512, gh: K=256, N=768 each), then GRU gate math -> h_new.
__global__ void k_gru(const int* __restrict__ src, const float* __restrict__ E_en,
                      const float* __restrict__ E_spa, const float* __restrict__ b_x,
                      const float* __restrict__ b_h,
                      const unsigned short* __restrict__ WcatT,
                      unsigned short* __restrict__ Acat, float* __restrict__ h,
                      unsigned short* __restrict__ hb16, float* __restrict__ gx_buf,
                      float* __restrict__ gh_buf, const int* __restrict__ st_wo,
                      const int* __restrict__ st_wr, const int* __restrict__ st_na,
                      const int* __restrict__ st_i) {
  __shared__ int toks[BATCH], tokt[BATCH];
  int tid = threadIdx.x;
  if (tid < BATCH) {
    int b = tid;
    int inp = src[st_i[b] * BATCH + b];
    if (st_wr[b]) inp = EN_NULL;
    if (st_na[b]) inp = EN_PAD;
    toks[b] = inp;
    tokt[b] = st_wo[b];
  }
  __syncthreads();
  // Phase A: A = [x_en | x_spa | h] (64 x 768) in bf16
  for (int e = tid; e < BATCH * 768; e += 256) {
    int b = e / 768, k = e % 768;
    float v;
    if (k < 256)      v = E_en[toks[b] * EMB + k];
    else if (k < 512) v = E_spa[tokt[b] * EMB + (k - 256)];
    else              v = h[b * HID + (k - 512)];
    Acat[e] = f2bf_rne(v);
  }
  __syncthreads();
  // Phase B: WMMA GEMMs. tiles 0..191 -> gx (K=512), 192..383 -> gh (K=256)
  int wave = tid >> 5, lane = tid & 31;
  int half = lane >> 4, lm = lane & 15;
  for (int t = wave; t < 384; t += 8) {
    bool is_gx = (t < 192);
    int tt = is_gx ? t : t - 192;
    int mt = tt & 3;           // 4 M-tiles (M=64)
    int nt = tt >> 2;          // 48 N-tiles (N=768)
    int m = mt * 16 + lm;
    int n = nt * 16 + lm;
    int k0 = is_gx ? 0 : 512;
    int ksteps = is_gx ? 16 : 8;
    const unsigned short* arow = Acat + m * 768 + k0;
    const unsigned short* brow = WcatT + n * 768 + k0;
    v8f c = {};
    for (int ks = 0; ks < ksteps; ++ks) {
      v16bf a = load_frag(arow + ks * 32, half);
      v16bf b = load_frag(brow + ks * 32, half);
      c = wmma_bf16(a, b, c);
    }
    float* dst = is_gx ? gx_buf : gh_buf;
#pragma unroll
    for (int r = 0; r < 8; ++r) {
      int mm = mt * 16 + (half ? 8 : 0) + r;
      dst[mm * 768 + n] = c[r];
    }
  }
  __syncthreads();
  // Phase C: GRU gates, h_new
  for (int e = tid; e < BATCH * HID; e += 256) {
    int b = e >> 8, cix = e & 255;
    float gxr = gx_buf[b * 768 + cix] + b_x[cix];
    float ghr = gh_buf[b * 768 + cix] + b_h[cix];
    float gxz = gx_buf[b * 768 + 256 + cix] + b_x[256 + cix];
    float ghz = gh_buf[b * 768 + 256 + cix] + b_h[256 + cix];
    float gxn = gx_buf[b * 768 + 512 + cix] + b_x[512 + cix];
    float ghn = gh_buf[b * 768 + 512 + cix] + b_h[512 + cix];
    float r = 1.0f / (1.0f + __expf(-(gxr + ghr)));
    float z = 1.0f / (1.0f + __expf(-(gxz + ghz)));
    float nn = tanhf(gxn + r * ghn);
    float hnew = (1.0f - z) * nn + z * h[e];
    h[e] = hnew;
    hb16[e] = f2bf_rne(hnew);
  }
}

// logits(64 x NPAD) = h_new @ W_out + b_out.
// Each workgroup: async-stage A (64x256 bf16) + a 32-row slab of WoutT into LDS,
// then 8 waves compute a 64x32 tile (wave -> (mt, nsel) = (wave>>1, wave&1)).
__global__ void k_out(const unsigned short* __restrict__ hb16,
                      const unsigned short* __restrict__ WoutT,
                      const float* __restrict__ b_out, float* __restrict__ logits) {
  extern __shared__ char smem[];
  const unsigned short* sA = (const unsigned short*)smem;
  const unsigned short* sB = (const unsigned short*)(smem + LDS_B_BASE);
  int tid = threadIdx.x;

  // stage A: 64 rows x 512B  (2048 x 16B chunks, 8 per thread)
  unsigned long long baseA = (unsigned long long)(uintptr_t)hb16;
#pragma unroll
  for (int i = 0; i < 8; ++i) {
    int c = tid + i * 256;
    int row = c >> 5;                 // 32 chunks per 512B row
    int col = (c & 31) << 4;
    async_b128((unsigned)(row * LDS_ROW_B + col),
               baseA + (unsigned)(row * 512 + col));
  }
  // stage B: WoutT rows [blockIdx*32, +32) x 512B (1024 chunks, 4 per thread)
  unsigned long long baseB = (unsigned long long)(uintptr_t)WoutT;
  size_t nbase = (size_t)blockIdx.x * 32;
#pragma unroll
  for (int i = 0; i < 4; ++i) {
    int c = tid + i * 256;
    int row = c >> 5;
    int col = (c & 31) << 4;
    if (nbase + (size_t)row < (size_t)NPAD) {
      async_b128((unsigned)(LDS_B_BASE + row * LDS_ROW_B + col),
                 baseB + (nbase + (size_t)row) * 512 + (unsigned)col);
    }
  }
  asm volatile("s_wait_asynccnt 0" ::: "memory");
  __syncthreads();

  int wave = tid >> 5, lane = tid & 31;
  int half = lane >> 4, lm = lane & 15;
  int mt = wave >> 1, nsel = wave & 1;
  int nt = blockIdx.x * 2 + nsel;
  if (nt >= NPAD / 16) return;        // wave-uniform (after the barrier)

  int m = mt * 16 + lm;
  int n = nt * 16 + lm;
  const unsigned short* arow = sA + m * LDS_ROW_E;
  const unsigned short* brow = sB + (nsel * 16 + lm) * LDS_ROW_E;
  v8f c = {};
#pragma unroll
  for (int ks = 0; ks < 8; ++ks) {
    v16bf a = load_frag(arow + ks * 32, half);
    v16bf b = load_frag(brow + ks * 32, half);
    c = wmma_bf16(a, b, c);
  }
  if (n < NOUT) {
    float bo = b_out[n];
#pragma unroll
    for (int r = 0; r < 8; ++r) {
      int mm = mt * 16 + (half ? 8 : 0) + r;
      logits[mm * NPAD + n] = c[r] + bo;
    }
  }
}

// per-row max / first-argmax / log-sum-exp over the 24000 vocab columns
__global__ void k_rowred(const float* __restrict__ logits, float* rowmax,
                         int* rowarg, float* rowlse) {
  int b = blockIdx.x, tid = threadIdx.x;
  __shared__ float sv[256];
  __shared__ int si[256];
  const float* row = logits + b * NPAD;
  float best = -3.4e38f; int bi = 0;
  for (int v = tid; v < SPA_V; v += 256) {
    float x = row[v];
    if (x > best) { best = x; bi = v; }
  }
  sv[tid] = best; si[tid] = bi;
  __syncthreads();
  for (int s = 128; s > 0; s >>= 1) {
    if (tid < s) {
      float o = sv[tid + s]; int oi = si[tid + s];
      if (o > sv[tid] || (o == sv[tid] && oi < si[tid])) { sv[tid] = o; si[tid] = oi; }
    }
    __syncthreads();
  }
  float m = sv[0]; int am = si[0];
  __syncthreads();
  float s = 0.0f;
  for (int v = tid; v < SPA_V; v += 256) s += __expf(row[v] - m);
  sv[tid] = s;
  __syncthreads();
  for (int st = 128; st > 0; st >>= 1) {
    if (tid < st) sv[tid] += sv[tid + st];
    __syncthreads();
  }
  if (tid == 0) { rowmax[b] = m; rowarg[b] = am; rowlse[b] = m + __logf(sv[0]); }
}

// state machine + Q_used; finalizes Q_target[t-1] using this step's qvals (nbav)
__global__ void k_state(int t, const int* __restrict__ src, const int* __restrict__ trg,
                        const int* __restrict__ eps, const int* __restrict__ rnd,
                        const int* __restrict__ tf, const float* __restrict__ logits,
                        const int* __restrict__ rowarg, const float* __restrict__ rowlse,
                        int* wo, int* wr, int* na, int* te, int* ia, int* ja,
                        int* outp, int* oldj_a, float* reward_p, int* reading_p,
                        int* naughty_p, int* justterm_p, int* term_p, int* counts,
                        float* q_used_out, float* q_target_out) {
  int b = threadIdx.x;
  if (b >= BATCH) return;
  const float* lrow = logits + b * NPAD;
  float q0 = lrow[SPA_V], q1 = lrow[SPA_V + 1], q2 = lrow[SPA_V + 2];
  float nbav = fmaxf(q0, fmaxf(q1, q2));
  if (t > 0) {  // finalize previous step's target (out2_{t-1} == out_t)
    float qt = reward_p[b] + 0.99f * nbav;
    if (term_p[b]) qt = 0.0f;
    if (reading_p[b]) qt = nbav;
    if (reading_p[b] && naughty_p[b]) qt = 0.99f * nbav;
    if (justterm_p[b]) qt = reward_p[b];
    qt -= 5.0f * (naughty_p[b] ? 1.0f : 0.0f);
    q_target_out[(t - 1) * BATCH + b] = qt;
  }
  int term_old = te[b];
  int amax = 0; float bv = q0;
  if (q1 > bv) { bv = q1; amax = 1; }
  if (q2 > bv) { bv = q2; amax = 2; }
  int action = (eps[t] > 0) ? rnd[t * BATCH + b] : amax;
  float qu = term_old ? 0.0f : ((action == 0) ? q0 : (action == 1) ? q1 : q2);
  q_used_out[t * BATCH + b] = qu;
  int reading = (!term_old) && (action == 0);
  int writing = (!term_old) && (action == 1);
  int bothing = (!term_old) && (action == 2);
  if (reading) atomicAdd(&counts[0], 1);
  if (writing) atomicAdd(&counts[1], 1);
  if (bothing) atomicAdd(&counts[2], 1);
  int outputting = writing || bothing;
  int old_i = ia[b], old_j = ja[b];
  int tgt = trg[old_j * BATCH + b];
  int just_term = outputting && (tgt == SPA_EOS);
  int naughty_n = (reading || bothing) && (src[old_i * BATCH + b] == EN_EOS);
  int i_n = old_i + ((!naughty_n && (reading || bothing)) ? 1 : 0);
  if (i_n > SRC_LEN - 1) i_n = SRC_LEN - 1;
  int j_n = old_j + (outputting ? 1 : 0);
  if (j_n > TRG_LEN - 1) j_n = TRG_LEN - 1;
  int term_n = term_old || just_term;
  int w = rowarg[b];
  if (tf[t] > 0) w = tgt;
  if (reading) w = SPA_NULL;
  float logp = lrow[tgt] - rowlse[b];
  float reward = (tgt == SPA_PAD) ? 0.0f : logp;
  reward_p[b] = reward; reading_p[b] = reading; naughty_p[b] = naughty_n;
  justterm_p[b] = just_term; term_p[b] = term_n;
  wo[b] = w; wr[b] = writing; na[b] = naughty_n; te[b] = term_n;
  ia[b] = i_n; ja[b] = j_n; outp[b] = outputting; oldj_a[b] = old_j;
}

// conditional copy of logits rows into word_outputs[old_j, b, :]
__global__ void k_copy(const float* __restrict__ logits, const int* __restrict__ outp,
                       const int* __restrict__ oldj, float* __restrict__ wout) {
  int b = blockIdx.y;
  if (!outp[b]) return;
  int v = blockIdx.x * 256 + threadIdx.x;
  if (v >= SPA_V) return;
  wout[((size_t)oldj[b] * BATCH + b) * SPA_V + v] = logits[b * NPAD + v];
}

// final pass: qvals of the 101st net_step (3 columns, f32), finalize Q_target[99], counts
__global__ void k_qfinal(const float* __restrict__ h, const float* __restrict__ Wout,
                         const float* __restrict__ b_out, const float* reward_p,
                         const int* reading_p, const int* naughty_p,
                         const int* justterm_p, const int* term_p, const int* counts,
                         float* q_target_out, float* counts_out) {
  int b = threadIdx.x;
  if (b < BATCH) {
    float a0 = b_out[SPA_V], a1 = b_out[SPA_V + 1], a2 = b_out[SPA_V + 2];
    for (int k = 0; k < HID; ++k) {
      float hv = h[b * HID + k];
      const float* wrow = Wout + k * NOUT + SPA_V;
      a0 += hv * wrow[0]; a1 += hv * wrow[1]; a2 += hv * wrow[2];
    }
    float nbav = fmaxf(a0, fmaxf(a1, a2));
    float qt = reward_p[b] + 0.99f * nbav;
    if (term_p[b]) qt = 0.0f;
    if (reading_p[b]) qt = nbav;
    if (reading_p[b] && naughty_p[b]) qt = 0.99f * nbav;
    if (justterm_p[b]) qt = reward_p[b];
    qt -= 5.0f * (naughty_p[b] ? 1.0f : 0.0f);
    q_target_out[(TSTEPS - 1) * BATCH + b] = qt;
  }
  if (b < 3) counts_out[b] = (float)counts[b];
}

// ---------------- host ----------------
extern "C" void kernel_launch(void* const* d_in, const int* in_sizes, int n_in,
                              void* d_out, int out_size, void* d_ws, size_t ws_size,
                              hipStream_t stream) {
  const int*   src   = (const int*)d_in[0];
  const int*   trg   = (const int*)d_in[1];
  const int*   eps   = (const int*)d_in[2];
  const int*   rnd   = (const int*)d_in[3];
  const int*   tf    = (const int*)d_in[4];
  const float* E_en  = (const float*)d_in[5];
  const float* E_spa = (const float*)d_in[6];
  const float* W_x   = (const float*)d_in[7];
  const float* W_h   = (const float*)d_in[8];
  const float* b_x   = (const float*)d_in[9];
  const float* b_h   = (const float*)d_in[10];
  const float* W_out = (const float*)d_in[11];
  const float* b_out = (const float*)d_in[12];

  float* out       = (float*)d_out;
  float* wout      = out;                                   // 50*64*24000
  float* q_used    = out + (size_t)TRG_LEN * BATCH * SPA_V; // +76800000
  float* q_target  = q_used + TSTEPS * BATCH;
  float* counts_o  = q_target + TSTEPS * BATCH;

  char* ws = (char*)d_ws;
  size_t off = 0;
  auto alloc = [&](size_t bytes) -> void* {
    void* p = ws + off;
    off = (off + bytes + 255) & ~(size_t)255;
    return p;
  };
  unsigned short* WcatT = (unsigned short*)alloc((size_t)768 * 768 * 2);
  unsigned short* WoutT = (unsigned short*)alloc((size_t)NPAD * 256 * 2);
  unsigned short* Acat  = (unsigned short*)alloc((size_t)BATCH * 768 * 2);
  unsigned short* hb16  = (unsigned short*)alloc((size_t)BATCH * HID * 2);
  float* h      = (float*)alloc((size_t)BATCH * HID * 4);
  float* gx_buf = (float*)alloc((size_t)BATCH * 768 * 4);
  float* gh_buf = (float*)alloc((size_t)BATCH * 768 * 4);
  float* logits = (float*)alloc((size_t)BATCH * NPAD * 4);
  float* rowmax = (float*)alloc(BATCH * 4);
  int*   rowarg = (int*)alloc(BATCH * 4);
  float* rowlse = (float*)alloc(BATCH * 4);
  int* wo   = (int*)alloc(BATCH * 4);
  int* wr   = (int*)alloc(BATCH * 4);
  int* na   = (int*)alloc(BATCH * 4);
  int* te   = (int*)alloc(BATCH * 4);
  int* ia   = (int*)alloc(BATCH * 4);
  int* ja   = (int*)alloc(BATCH * 4);
  int* outp = (int*)alloc(BATCH * 4);
  int* oldj = (int*)alloc(BATCH * 4);
  float* reward_p  = (float*)alloc(BATCH * 4);
  int* reading_p   = (int*)alloc(BATCH * 4);
  int* naughty_p   = (int*)alloc(BATCH * 4);
  int* justterm_p  = (int*)alloc(BATCH * 4);
  int* term_p      = (int*)alloc(BATCH * 4);
  int* counts      = (int*)alloc(8 * 4);

  const int wo_elems = TRG_LEN * BATCH * SPA_V;
  k_zero<<<(wo_elems + 255) / 256, 256, 0, stream>>>(wout, wo_elems);
  k_init<<<(BATCH * HID + 255) / 256, 256, 0, stream>>>(h, wo, wr, na, te, ia, ja, counts);
  k_convW<<<(768 * 768 + 255) / 256, 256, 0, stream>>>(W_x, W_h, WcatT);
  k_convWout<<<(NPAD * 256 + 255) / 256, 256, 0, stream>>>(W_out, WoutT);

  const int out_blocks = (NPAD / 16 + 1) / 2;  // 751 blocks, 2 N-tiles each
  for (int t = 0; t < TSTEPS; ++t) {
    k_gru<<<1, 256, 0, stream>>>(src, E_en, E_spa, b_x, b_h, WcatT, Acat, h, hb16,
                                 gx_buf, gh_buf, wo, wr, na, ia);
    k_out<<<out_blocks, 256, LDS_TOTAL, stream>>>(hb16, WoutT, b_out, logits);
    k_rowred<<<BATCH, 256, 0, stream>>>(logits, rowmax, rowarg, rowlse);
    k_state<<<1, 64, 0, stream>>>(t, src, trg, eps, rnd, tf, logits, rowarg, rowlse,
                                  wo, wr, na, te, ia, ja, outp, oldj, reward_p,
                                  reading_p, naughty_p, justterm_p, term_p, counts,
                                  q_used, q_target);
    k_copy<<<dim3((SPA_V + 255) / 256, BATCH), 256, 0, stream>>>(logits, outp, oldj, wout);
  }
  // 101st net_step: supplies nbav for step 99
  k_gru<<<1, 256, 0, stream>>>(src, E_en, E_spa, b_x, b_h, WcatT, Acat, h, hb16,
                               gx_buf, gh_buf, wo, wr, na, ia);
  k_qfinal<<<1, 64, 0, stream>>>(h, W_out, b_out, reward_p, reading_p, naughty_p,
                                 justterm_p, term_p, counts, q_target, counts_o);
}
